// InvertedResidualBlockCarafe_80041010528290
// MI455X (gfx1250) — compile-verified, matrix operand order checked by reading, not run on hardware
//
#include <hip/hip_runtime.h>

// ---------------------------------------------------------------------------
// InvertedResidualBlockCarafe for MI455X (gfx1250, wave32, WMMA bf16)
//
// Activations are kept CHANNEL-LAST ([b, n, k], k contiguous) so every WMMA
// B-fragment is two 16-byte global_load_b128 per lane (K runs contiguous),
// mirroring the A-fragment weight loads.
//
//   K0: xt  = transpose(x) -> bf16 [b,n,64]     (LDS tile transpose)
//   K1: h1t = relu6(bn1(w1 @ x))   GEMM M=384 K=64  -> bf16 [b,n,384]
//   K2: y   = wc @ h1 + bc         GEMM M=64  K=384 -> f32  [b,m,n]
//   K3: wgt = softmax_c(conv5x5_s2(y))           VALU, LDS-staged weights
//   K4: h2t = relu6(bn2(carafe(h1t,wgt)))        VALU -> bf16 [b,n,384]
//   K5: out = bn3(w2 @ h2)         GEMM M=128 K=384 -> f32  [b,m,n]
// ---------------------------------------------------------------------------

typedef __attribute__((ext_vector_type(16))) __bf16 v16bf;
typedef __attribute__((ext_vector_type(8)))  __bf16 v8bf;
typedef __attribute__((ext_vector_type(8)))  float  v8f;

// CDNA5 ISA 7.12.2: 16-bit A/B fragment, vector element e -> K index
static __device__ __forceinline__ int wmma_k(int e, int half) {
    return (e & 7) + (half << 3) + ((e >> 3) << 4);
}

// A fragment: lane = row, 16 f32 weights at wr[k0 .. k0+31] (two 32B runs,
// compiler merges to global_load_b128 pairs)
static __device__ __forceinline__ v16bf load_a_f32(const float* __restrict__ wr,
                                                   int k0, int half) {
    v16bf a;
    #pragma unroll
    for (int e = 0; e < 16; ++e)
        a[e] = (__bf16)wr[k0 + wmma_k(e, half)];
    return a;
}

// B fragment from channel-last bf16 activations: two 16B packed loads.
// p points at the lane's column (n fixed); element k contiguous.
static __device__ __forceinline__ v16bf load_b_packed(const __bf16* __restrict__ p,
                                                      int k0, int half) {
    const v8bf lo = *(const v8bf*)(p + k0 + 8 * half);
    const v8bf hi = *(const v8bf*)(p + k0 + 16 + 8 * half);
    v16bf f;
    #pragma unroll
    for (int e = 0; e < 8; ++e) { f[e] = lo[e]; f[e + 8] = hi[e]; }
    return f;
}

#define WMMA_BF16(A, B, C) \
    __builtin_amdgcn_wmma_f32_16x16x32_bf16(false, (A), false, (B), (short)0, (C), false, false)

// ---------------------------------------------------------------------------
// K0: xt[b,n,c] (bf16) = x[b,c,n] (f32).  64c x 64n tile via LDS.
// 8 batches * 196 n-chunks = 1568 blocks of 256 threads.
// ---------------------------------------------------------------------------
__global__ void __launch_bounds__(256) k_transpose_x(
    const float* __restrict__ x, __bf16* __restrict__ xt)
{
    __shared__ float sT[64][65];                 // pad -> conflict-free
    const int blk = blockIdx.x;
    const int b = blk / 196, nc = blk % 196;
    const int n0 = nc * 64;
    const int tid = threadIdx.x;

    const int cr = tid >> 6, nn = tid & 63;      // read: 4 c-rows / pass
    #pragma unroll
    for (int p = 0; p < 16; ++p) {
        const int c = p * 4 + cr;
        sT[c][nn] = x[((size_t)b * 64 + c) * 12544 + n0 + nn];
    }
    __syncthreads();
    const int cc = tid & 63, nr = tid >> 6;      // write: c contiguous
    #pragma unroll
    for (int p = 0; p < 16; ++p) {
        const int n = p * 4 + nr;
        xt[((size_t)b * 12544 + n0 + n) * 64 + cc] = (__bf16)sT[cc][n];
    }
}

// ---------------------------------------------------------------------------
// K1: h1t[b,n,m] = relu6(bn1( sum_k w1[m,k] * xt[b,n,k] ))   n = h*112+w
// M=384 (24 tiles), N=12544 (196 groups of 64), K=64 (2 steps). 8 WMMA/wave.
// ---------------------------------------------------------------------------
__global__ void __launch_bounds__(256) k_gemm1(
    const __bf16* __restrict__ xt, const float* __restrict__ w1,
    const float* __restrict__ g1,  const float* __restrict__ b1,
    const float* __restrict__ m1,  const float* __restrict__ v1,
    __bf16* __restrict__ h1t)
{
    const int lane = threadIdx.x & 31;
    const int gw   = blockIdx.x * 8 + (threadIdx.x >> 5);
    const int NG = 196, MT = 24;
    const int b   = gw / (MT * NG);
    const int rem = gw % (MT * NG);
    const int mt = rem / NG, ng = rem % NG;
    const int half = lane >> 4, lr = lane & 15;
    const int m0 = mt * 16, n0 = ng * 64 + lr;

    const float* w1r = w1 + (size_t)(m0 + lr) * 64;
    const __bf16* xb[4];
    #pragma unroll
    for (int t = 0; t < 4; ++t)
        xb[t] = xt + ((size_t)b * 12544 + n0 + t * 16) * 64;

    v8f acc[4] = {};
    #pragma unroll
    for (int kk = 0; kk < 2; ++kk) {
        const v16bf a = load_a_f32(w1r, kk * 32, half);
        #pragma unroll
        for (int t = 0; t < 4; ++t) {
            const v16bf bm = load_b_packed(xb[t], kk * 32, half);
            acc[t] = WMMA_BF16(a, bm, acc[t]);
        }
    }
    float inv[8], sh[8];
    #pragma unroll
    for (int v = 0; v < 8; ++v) {
        const int m = m0 + v + half * 8;
        inv[v] = g1[m] * rsqrtf(v1[m] + 1e-5f);
        sh[v]  = b1[m] - m1[m] * inv[v];
    }
    #pragma unroll
    for (int t = 0; t < 4; ++t) {
        v8bf pk;
        #pragma unroll
        for (int v = 0; v < 8; ++v) {
            float val = acc[t][v] * inv[v] + sh[v];
            val = fminf(fmaxf(val, 0.f), 6.f);
            pk[v] = (__bf16)val;
        }
        *(v8bf*)(h1t + ((size_t)b * 12544 + n0 + t * 16) * 384 + m0 + 8 * half) = pk;
    }
}

// ---------------------------------------------------------------------------
// K2: y[b,m,n] = sum_k wc[m,k] * h1t[b,n,k] + bc[m]
// M=64 (4 tiles), N=196 groups, K=384 (12 steps). 48 WMMA/wave.
// ---------------------------------------------------------------------------
__global__ void __launch_bounds__(256) k_gemm2(
    const __bf16* __restrict__ h1t, const float* __restrict__ wc,
    const float* __restrict__ bc,   float* __restrict__ y)
{
    const int lane = threadIdx.x & 31;
    const int gw   = blockIdx.x * 8 + (threadIdx.x >> 5);
    const int NG = 196, MT = 4;
    const int b   = gw / (MT * NG);
    const int rem = gw % (MT * NG);
    const int mt = rem / NG, ng = rem % NG;
    const int half = lane >> 4, lr = lane & 15;
    const int m0 = mt * 16, n0 = ng * 64 + lr;

    const float* wr = wc + (size_t)(m0 + lr) * 384;
    const __bf16* hb[4];
    #pragma unroll
    for (int t = 0; t < 4; ++t)
        hb[t] = h1t + ((size_t)b * 12544 + n0 + t * 16) * 384;

    v8f acc[4] = {};
    for (int kk = 0; kk < 12; ++kk) {
        const v16bf a = load_a_f32(wr, kk * 32, half);
        #pragma unroll
        for (int t = 0; t < 4; ++t) {
            const v16bf bm = load_b_packed(hb[t], kk * 32, half);
            acc[t] = WMMA_BF16(a, bm, acc[t]);
        }
    }
    float* op = y + (size_t)b * 64 * 12544 + n0;
    #pragma unroll
    for (int v = 0; v < 8; ++v) {
        const int m = m0 + v + half * 8;
        const float bias = bc[m];
        #pragma unroll
        for (int t = 0; t < 4; ++t)
            op[(size_t)m * 12544 + t * 16] = acc[t][v] + bias;
    }
}

// ---------------------------------------------------------------------------
// K3: 5x5 stride-2 pad-2 conv (64 -> 25 ch) + channel softmax.
// One thread per output pixel (8*56*56 = 25088 = 98 blocks * 256 exactly).
// Per-tap weight slice (25*64 f32 = 6.25 KB) staged in LDS; all lanes read
// the same LDS word -> conflict-free broadcast.
// ---------------------------------------------------------------------------
__global__ void __launch_bounds__(256) k_conv5_softmax(
    const float* __restrict__ y, const float* __restrict__ we,
    const float* __restrict__ be, float* __restrict__ wgt)
{
    __shared__ float sW[1600];
    const int tid = threadIdx.x;
    const int p   = blockIdx.x * 256 + tid;
    const int b   = p / 3136;
    const int rem = p % 3136;
    const int ho = rem / 56, wo = rem % 56;

    float e[25];
    #pragma unroll
    for (int o = 0; o < 25; ++o) e[o] = be[o];

    for (int t = 0; t < 25; ++t) {
        __syncthreads();
        for (int i = tid; i < 1600; i += 256) {
            const int o = i >> 6, cm = i & 63;
            sW[i] = we[o * 1600 + cm * 25 + t];   // we[o][cm][di][dj], tap t
        }
        __syncthreads();
        const int di = t / 5, dj = t % 5;
        const int hi = 2 * ho + di - 2, wi = 2 * wo + dj - 2;
        if ((unsigned)hi < 112u && (unsigned)wi < 112u) {
            const float* yp = y + (size_t)b * 64 * 12544 + hi * 112 + wi;
            for (int cm = 0; cm < 64; ++cm) {
                const float yv = yp[(size_t)cm * 12544];
                #pragma unroll
                for (int o = 0; o < 25; ++o)
                    e[o] = fmaf(yv, sW[o * 64 + cm], e[o]);
            }
        }
    }
    float mx = e[0];
    #pragma unroll
    for (int o = 1; o < 25; ++o) mx = fmaxf(mx, e[o]);
    float s = 0.f;
    #pragma unroll
    for (int o = 0; o < 25; ++o) { e[o] = __expf(e[o] - mx); s += e[o]; }
    const float inv = 1.f / s;
    float* wp = wgt + (size_t)b * 25 * 3136 + ho * 56 + wo;
    #pragma unroll
    for (int o = 0; o < 25; ++o) wp[(size_t)o * 3136] = e[o] * inv;
}

// ---------------------------------------------------------------------------
// K4: h2t[b,n,c] = relu6(bn2( sum_tap h1t[b,2ho+di-2:2wo+dj-2,c]*wgt[b,tap,n] ))
// c-fastest threads: contiguous h1t reads, wave-uniform wgt reads, and the
// output index equals the flat thread index.  9 633 792 = 37632 * 256.
// ---------------------------------------------------------------------------
__global__ void __launch_bounds__(256) k_carafe_bn2(
    const __bf16* __restrict__ h1t, const float* __restrict__ wgt,
    const float* __restrict__ g2,   const float* __restrict__ b2,
    const float* __restrict__ m2,   const float* __restrict__ v2,
    __bf16* __restrict__ h2t)
{
    const size_t idx = (size_t)blockIdx.x * 256 + threadIdx.x;
    const int c = (int)(idx % 384);
    const int n = (int)((idx / 384) % 3136);
    const int b = (int)(idx / (384 * 3136));
    const int ho = n / 56, wo = n % 56;

    const __bf16* hp = h1t + (size_t)b * 12544 * 384 + c;
    const float*  wp = wgt + (size_t)b * 25 * 3136 + n;

    float acc = 0.f;
    #pragma unroll
    for (int di = 0; di < 5; ++di) {
        const int hi = 2 * ho + di - 2;
        if ((unsigned)hi >= 112u) continue;      // zero padding
        #pragma unroll
        for (int dj = 0; dj < 5; ++dj) {
            const int wi = 2 * wo + dj - 2;
            if ((unsigned)wi >= 112u) continue;
            acc = fmaf((float)hp[(size_t)(hi * 112 + wi) * 384],
                       wp[(size_t)(di * 5 + dj) * 3136], acc);
        }
    }
    const float inv = g2[c] * rsqrtf(v2[c] + 1e-5f);
    float val = acc * inv + (b2[c] - m2[c] * inv);
    val = fminf(fmaxf(val, 0.f), 6.f);
    h2t[idx] = (__bf16)val;
}

// ---------------------------------------------------------------------------
// K5: out[b,m,n] = bn3( sum_k w2[m,k] * h2t[b,n,k] )   n = ho*56+wo
// M=128 (8 tiles), N=3136 (49 groups of 64), K=384 (12 steps). 48 WMMA/wave.
// ---------------------------------------------------------------------------
__global__ void __launch_bounds__(256) k_gemm3(
    const __bf16* __restrict__ h2t, const float* __restrict__ w2,
    const float* __restrict__ g3,   const float* __restrict__ b3,
    const float* __restrict__ m3,   const float* __restrict__ v3,
    float* __restrict__ out)
{
    const int lane = threadIdx.x & 31;
    const int gw   = blockIdx.x * 8 + (threadIdx.x >> 5);
    const int NG = 49, MT = 8;
    const int b   = gw / (MT * NG);
    const int rem = gw % (MT * NG);
    const int mt = rem / NG, ng = rem % NG;
    const int half = lane >> 4, lr = lane & 15;
    const int m0 = mt * 16, n0 = ng * 64 + lr;

    const float* wr = w2 + (size_t)(m0 + lr) * 384;
    const __bf16* hb[4];
    #pragma unroll
    for (int t = 0; t < 4; ++t)
        hb[t] = h2t + ((size_t)b * 3136 + n0 + t * 16) * 384;

    v8f acc[4] = {};
    for (int kk = 0; kk < 12; ++kk) {
        const v16bf a = load_a_f32(wr, kk * 32, half);
        #pragma unroll
        for (int t = 0; t < 4; ++t) {
            const v16bf bm = load_b_packed(hb[t], kk * 32, half);
            acc[t] = WMMA_BF16(a, bm, acc[t]);
        }
    }
    float* op = out + (size_t)b * 128 * 3136 + n0;
    #pragma unroll
    for (int v = 0; v < 8; ++v) {
        const int m = m0 + v + half * 8;
        const float inv = g3[m] * rsqrtf(v3[m] + 1e-5f);
        const float sh  = b3[m] - m3[m] * inv;
        #pragma unroll
        for (int t = 0; t < 4; ++t)
            op[(size_t)m * 3136 + t * 16] = acc[t][v] * inv + sh;
    }
}

// ---------------------------------------------------------------------------
// Workspace layout (bytes, 256-aligned):
//   h1t bf16  8*12544*384*2 = 77 070 336   @ 0
//   xt  bf16  8*12544*64 *2 = 12 845 056   @ 77 070 336
//   y   f32   8*64*12544 *4 = 25 690 112   @ 89 915 392
//   wgt f32   8*25*3136  *4 =  2 508 800   @ 115 605 504
//   h2t bf16  8*3136*384 *2 = 19 267 584   @ 118 114 304
// total 137 381 888 bytes
// ---------------------------------------------------------------------------
extern "C" void kernel_launch(void* const* d_in, const int* in_sizes, int n_in,
                              void* d_out, int out_size, void* d_ws, size_t ws_size,
                              hipStream_t stream)
{
    (void)in_sizes; (void)n_in; (void)out_size; (void)ws_size;

    const float* x  = (const float*)d_in[0];
    const float* w1 = (const float*)d_in[1];
    const float* g1 = (const float*)d_in[2];
    const float* b1 = (const float*)d_in[3];
    const float* m1 = (const float*)d_in[4];
    const float* v1 = (const float*)d_in[5];
    const float* wc = (const float*)d_in[6];
    const float* bc = (const float*)d_in[7];
    const float* we = (const float*)d_in[8];
    const float* be = (const float*)d_in[9];
    const float* g2 = (const float*)d_in[10];
    const float* b2 = (const float*)d_in[11];
    const float* m2 = (const float*)d_in[12];
    const float* v2 = (const float*)d_in[13];
    const float* w2 = (const float*)d_in[14];
    const float* g3 = (const float*)d_in[15];
    const float* b3 = (const float*)d_in[16];
    const float* m3 = (const float*)d_in[17];
    const float* v3 = (const float*)d_in[18];

    char* ws = (char*)d_ws;
    __bf16* h1t = (__bf16*)(ws);
    __bf16* xt  = (__bf16*)(ws + 77070336ull);
    float*  yb  = (float*) (ws + 89915392ull);
    float*  wgt = (float*) (ws + 115605504ull);
    __bf16* h2t = (__bf16*)(ws + 118114304ull);
    float*  out = (float*)d_out;

    // K0: 8 * 196 tiles
    k_transpose_x<<<1568, 256, 0, stream>>>(x, xt);
    // K1: 8 * 24 * 196 = 37632 waves / 8 per block
    k_gemm1<<<4704, 256, 0, stream>>>(xt, w1, g1, b1, m1, v1, h1t);
    // K2: 8 * 4 * 196 = 6272 waves
    k_gemm2<<<784, 256, 0, stream>>>(h1t, wc, bc, yb);
    // K3: 25088 pixels
    k_conv5_softmax<<<98, 256, 0, stream>>>(yb, we, be, wgt);
    // K4: 9 633 792 elements
    k_carafe_bn2<<<37632, 256, 0, stream>>>(h1t, wgt, g2, b2, m2, v2, h2t);
    // K5: 8 * 8 * 49 = 3136 waves
    k_gemm3<<<392, 256, 0, stream>>>(h2t, w2, g3, b3, m3, v3, out);
}
